// LSTMLanguageModelScratch_13082470383726
// MI455X (gfx1250) — matrix operand order repaired and checked by reading, not compiled, not run on hardware
//
#include <hip/hip_runtime.h>
#include <hip/hip_bf16.h>

typedef __attribute__((ext_vector_type(16))) _Float16 v16h;
typedef __attribute__((ext_vector_type(8)))  _Float16 v8h;
typedef __attribute__((ext_vector_type(8)))  float    v8f;

#define VOCAB 8000
#define EMB   128
#define HID   256
#define BATCH 32
#define SEQ   512
#define KDIM  (EMB + HID)     // 384
#define NWG   4               // cooperating workgroups in LSTM recurrence
#define HSL   (HID / NWG)     // 64 hidden units per WG
#define GCOLS (4 * HSL)       // 256 gate columns per WG (f|i|o|g blocked)

__device__ __forceinline__ float sigmoidf_(float x) {
    return 1.0f / (1.0f + __expf(-x));
}

// A-operand swizzle: within each 32-element (16-bit) group, swap 8-half blocks
// 1 and 2. Lane g then reads its two WMMA K-runs as ONE contiguous 32B span:
//   elements 0..7  -> K = 32t + 8g .. +7   (VGPRs 0-3)
//   elements 8..15 -> K = 32t + 16 + 8g .. (VGPRs 4-7)
__device__ __forceinline__ int swz16(int k) {
    int j  = (k >> 3) & 3;
    int jp = ((j & 1) << 1) | (j >> 1);
    return (k & ~31) + jp * 8 + (k & 7);
}

// LDS byte offset of a __shared__ pointer (flat LDS addr low 32 bits).
__device__ __forceinline__ unsigned lds_off(const void* p) {
    return (unsigned)(size_t)p;
}

__device__ __forceinline__ void async_copy16(unsigned lds_addr,
                                             unsigned long long gaddr) {
    asm volatile("global_load_async_to_lds_b128 %0, %1, off"
                 :: "v"(lds_addr), "v"(gaddr) : "memory");
}

__device__ __forceinline__ void wait_async0() {
    asm volatile("s_wait_asynccnt 0x0" ::: "memory");
}

// ---------------------------------------------------------------------------
// Embedding gather (swizzled, time-major): et[s][b][swz(e)] = f16(emb[x[b][s]][e])
// ---------------------------------------------------------------------------
__global__ void k_embed(const long long* __restrict__ x,
                        const float* __restrict__ emb,
                        _Float16* __restrict__ et) {
    int rid = blockIdx.x;                 // 0 .. SEQ*BATCH-1
    int s = rid / BATCH, b = rid % BATCH;
    long long xi = x[(size_t)b * SEQ + s];
    const float* src = emb + (size_t)xi * EMB;
    _Float16* dst = et + ((size_t)s * BATCH + b) * EMB;
    int t = threadIdx.x;
    dst[swz16(t)] = (_Float16)src[t];
}

// ---------------------------------------------------------------------------
// Weight packing (grid-stride):
//   WpackT[wg][c][k] = f16(W_{gate}[k][wg*64 + c%64]),  gate = c/64
//   WoutT[v][h]      = f16(Wout[h][v])
//   bpack, hzero, counter init
// ---------------------------------------------------------------------------
__global__ void k_pack(const float* __restrict__ Wf, const float* __restrict__ Wi,
                       const float* __restrict__ Wo, const float* __restrict__ Wg,
                       const float* __restrict__ bf, const float* __restrict__ bi,
                       const float* __restrict__ bo, const float* __restrict__ bg,
                       const float* __restrict__ Wout,
                       _Float16* __restrict__ WpackT, float* __restrict__ bpack,
                       _Float16* __restrict__ WoutT, _Float16* __restrict__ hzero,
                       unsigned int* __restrict__ counter) {
    const float* W4[4] = {Wf, Wi, Wo, Wg};
    const float* b4[4] = {bf, bi, bo, bg};
    const size_t NW = (size_t)NWG * GCOLS * KDIM;
    const size_t NO = (size_t)VOCAB * HID;
    const size_t NB = (size_t)NWG * GCOLS;
    const size_t NZ = (size_t)BATCH * HID;
    const size_t total = NW + NO + NB + NZ + 1;
    for (size_t i = (size_t)blockIdx.x * blockDim.x + threadIdx.x; i < total;
         i += (size_t)gridDim.x * blockDim.x) {
        if (i < NW) {
            size_t t = i;
            int k = (int)(t % KDIM); t /= KDIM;
            int c = (int)(t % GCOLS);
            int wg = (int)(t / GCOLS);
            int gate = c >> 6;
            int u = wg * HSL + (c & 63);
            WpackT[i] = (_Float16)W4[gate][(size_t)k * HID + u];
        } else if (i < NW + NO) {
            size_t t = i - NW;
            int h = (int)(t % HID);
            size_t v = t / HID;
            WoutT[t] = (_Float16)Wout[(size_t)h * VOCAB + v];
        } else if (i < NW + NO + NB) {
            size_t t = i - NW - NO;
            int c = (int)(t % GCOLS);
            int wg = (int)(t / GCOLS);
            int gate = c >> 6;
            int u = wg * HSL + (c & 63);
            bpack[t] = b4[gate][u];
        } else if (i < NW + NO + NB + NZ) {
            hzero[i - NW - NO - NB] = (_Float16)0.0f;
        } else {
            *counter = 0u;
        }
    }
}

// ---------------------------------------------------------------------------
// LSTM recurrence: NWG workgroups, each owns 64 hidden units (256 gate cols).
// Weight slice async-staged into LDS (192 KB). Per step: [32x384]@[384x256]
// via WMMA, activations, c/h update, h published (swizzled) to hs[s],
// atomic-counter barrier across the NWG co-resident workgroups.
// ---------------------------------------------------------------------------
__global__ void __launch_bounds__(256, 1)
k_lstm(const _Float16* __restrict__ WpackT, const float* __restrict__ bpack,
       const _Float16* __restrict__ et, const _Float16* __restrict__ hzero,
       _Float16* __restrict__ hs, unsigned int* __restrict__ counter) {
    __shared__ _Float16 sW[GCOLS * KDIM];   // 192 KB: [col][k], k contiguous
    __shared__ float    sbias[GCOLS];
    __shared__ float    sc[BATCH * HSL];    // cell state slice, 8 KB
    __shared__ float    sg[4][BATCH * HSL]; // f,i,o,g activations, 32 KB

    const int tid = threadIdx.x;
    const int wg  = blockIdx.x;

    // Async-stage this WG's weight slice (192 KB) into LDS.
    {
        const char* src = (const char*)(WpackT + (size_t)wg * GCOLS * KDIM);
        unsigned ldsW = lds_off(sW);
        for (int c = tid; c < GCOLS * KDIM * 2 / 16; c += 256)
            async_copy16(ldsW + c * 16,
                         (unsigned long long)(src + (size_t)c * 16));
        sbias[tid] = bpack[wg * GCOLS + tid];
    }
    for (int r = tid; r < BATCH * HSL; r += 256) sc[r] = 0.0f;
    wait_async0();
    __syncthreads();

    const int lane  = tid & 31;
    const int w     = tid >> 5;          // wave 0..7
    const int g     = lane >> 4;         // half-wave group
    const int l15   = lane & 15;
    const int mtile = w & 1;             // batch-rows tile
    const int nbase = (w >> 1) * 64;     // this wave's gate-column block
    const int gate  = w >> 1;            // 0=f 1=i 2=o 3=g
    const int mrow  = mtile * 16 + l15;  // A-fragment batch row for this lane

    for (int s = 0; s < SEQ; ++s) {
        const _Float16* hprev = (s == 0) ? hzero : (hs + (size_t)(s - 1) * BATCH * HID);
        const _Float16* xrow  = et + (size_t)s * BATCH * EMB;

        v8f acc[4];
        #pragma unroll
        for (int nt = 0; nt < 4; ++nt) {
            float bv = sbias[nbase + nt * 16 + l15];
            #pragma unroll
            for (int j = 0; j < 8; ++j) acc[nt][j] = bv;
        }

        #pragma unroll
        for (int kt = 0; kt < KDIM / 32; ++kt) {
            // A fragment: single contiguous 32B span (sources pre-swizzled).
            const _Float16* abase;
            if (kt < HID / 32)
                abase = hprev + (size_t)mrow * HID + kt * 32 + g * 16;
            else
                abase = xrow + (size_t)mrow * EMB + (kt - HID / 32) * 32 + g * 16;
            v16h A = *(const v16h*)abase;

            #pragma unroll
            for (int nt = 0; nt < 4; ++nt) {
                const _Float16* bp =
                    sW + (size_t)(nbase + nt * 16 + l15) * KDIM + kt * 32 + g * 8;
                v8h blo = *(const v8h*)bp;    // two 16B K-runs (table order) for column n
                v8h bhi = *(const v8h*)(bp + 16);
                v16h Bm = __builtin_shufflevector(blo, bhi, 0, 1, 2, 3, 4, 5, 6, 7, 8, 9, 10, 11, 12, 13, 14, 15);
                acc[nt] = __builtin_amdgcn_wmma_f32_16x16x32_f16(
                    false, A, false, Bm, (short)0, acc[nt], false, false);
            }
        }

        // Activations -> LDS gate buffers (each wave covers one gate, 16 rows).
        #pragma unroll
        for (int nt = 0; nt < 4; ++nt) {
            int ul = nt * 16 + l15;               // local hidden unit 0..63
            #pragma unroll
            for (int j = 0; j < 8; ++j) {
                int m = mtile * 16 + g * 8 + j;   // batch row
                float v = acc[nt][j];
                float a = (gate == 3) ? tanhf(v) : sigmoidf_(v);
                sg[gate][m * HSL + ul] = a;
            }
        }
        __syncthreads();

        // c/h update for this WG's 32x64 slice; publish swizzled h to hs[s].
        for (int r = tid; r < BATCH * HSL; r += 256) {
            int b_ = r / HSL, u = r % HSL;
            float f = sg[0][r], ii = sg[1][r], o = sg[2][r], gg = sg[3][r];
            float c = f * sc[r] + ii * gg;
            sc[r] = c;
            float h = o * tanhf(c);
            hs[(size_t)s * BATCH * HID + (size_t)b_ * HID + swz16(wg * HSL + u)] =
                (_Float16)h;
        }
        __threadfence();
        __syncthreads();
        if (tid == 0) {
            __hip_atomic_fetch_add(counter, 1u, __ATOMIC_ACQ_REL,
                                   __HIP_MEMORY_SCOPE_AGENT);
            unsigned target = (unsigned)NWG * (unsigned)(s + 1);
            while (__hip_atomic_load(counter, __ATOMIC_ACQUIRE,
                                     __HIP_MEMORY_SCOPE_AGENT) < target)
                __builtin_amdgcn_s_sleep(1);
        }
        __syncthreads();
    }
}

// ---------------------------------------------------------------------------
// Output projection: [S*B x 256] @ [256 x 8000] + bout, f16 WMMA / f32 acc.
// Grid: (128 M-blocks of 128 rows, 125 N-blocks of 64 cols).
// A block (64 KB) + B block (32 KB) async-staged into LDS; main loop is pure
// LDS reads + WMMA; non-temporal f32 stores for the 524 MB output.
// ---------------------------------------------------------------------------
__global__ void __launch_bounds__(256)
k_proj(const _Float16* __restrict__ hs, const _Float16* __restrict__ WoutT,
       const float* __restrict__ bout, float* __restrict__ out) {
    __shared__ _Float16 sA[128 * HID];   // 64 KB  [local row][k] (pre-swizzled)
    __shared__ _Float16 sB[64 * HID];    // 32 KB  [local col][k]

    const int tid   = threadIdx.x;
    const int mbase = blockIdx.x * 128;
    const int nbase = blockIdx.y * 64;

    {
        const char* gA = (const char*)(hs + (size_t)mbase * HID);
        unsigned ldsA = lds_off(sA);
        #pragma unroll 4
        for (int c = tid; c < 4096; c += 256)     // 128 rows * 512 B
            async_copy16(ldsA + c * 16,
                         (unsigned long long)(gA + (size_t)c * 16));
        const char* gB = (const char*)(WoutT + (size_t)nbase * HID);
        unsigned ldsB = lds_off(sB);
        #pragma unroll 4
        for (int c = tid; c < 2048; c += 256)     // 64 cols * 512 B
            async_copy16(ldsB + c * 16,
                         (unsigned long long)(gB + (size_t)c * 16));
    }

    const int lane = tid & 31;
    const int w    = tid >> 5;
    const int g    = lane >> 4;
    const int l15  = lane & 15;
    const int m0L  = w * 16;             // local 16-row tile base

    v8f acc[4];
    #pragma unroll
    for (int nt = 0; nt < 4; ++nt) {
        float bv = bout[nbase + nt * 16 + l15];
        #pragma unroll
        for (int j = 0; j < 8; ++j) acc[nt][j] = bv;
    }

    wait_async0();
    __syncthreads();

    const _Float16* arow = sA + (size_t)(m0L + l15) * HID;
    #pragma unroll
    for (int kt = 0; kt < HID / 32; ++kt) {
        v16h A = *(const v16h*)(arow + kt * 32 + g * 16);   // swizzled: one load
        #pragma unroll
        for (int nt = 0; nt < 4; ++nt) {
            const _Float16* bp =
                sB + (size_t)(nt * 16 + l15) * HID + kt * 32 + g * 8;
            v8h blo = *(const v8h*)bp;
            v8h bhi = *(const v8h*)(bp + 16);
            v16h Bm = __builtin_shufflevector(blo, bhi, 0, 1, 2, 3, 4, 5, 6, 7, 8, 9, 10, 11, 12, 13, 14, 15);
            acc[nt] = __builtin_amdgcn_wmma_f32_16x16x32_f16(
                false, A, false, Bm, (short)0, acc[nt], false, false);
        }
    }

    // Store: D lane holds col n = nbase+nt*16+l15, rows (mbase+m0L) + g*8 + j.
    #pragma unroll
    for (int nt = 0; nt < 4; ++nt) {
        int n = nbase + nt * 16 + l15;
        #pragma unroll
        for (int j = 0; j < 8; ++j) {
            int m = mbase + m0L + g * 8 + j;
            int s = m >> 5, b = m & 31;   // m = s*32 + b
            __builtin_nontemporal_store(acc[nt][j],
                out + ((size_t)b * SEQ + s) * VOCAB + n);
        }
    }
}

// ---------------------------------------------------------------------------
extern "C" void kernel_launch(void* const* d_in, const int* in_sizes, int n_in,
                              void* d_out, int out_size, void* d_ws, size_t ws_size,
                              hipStream_t stream) {
    (void)in_sizes; (void)n_in; (void)out_size; (void)ws_size;
    const long long* x   = (const long long*)d_in[0];
    const float* emb     = (const float*)d_in[1];
    const float* Wf      = (const float*)d_in[2];
    const float* bf      = (const float*)d_in[3];
    const float* Wi      = (const float*)d_in[4];
    const float* bi      = (const float*)d_in[5];
    const float* Wo      = (const float*)d_in[6];
    const float* bo      = (const float*)d_in[7];
    const float* Wg      = (const float*)d_in[8];
    const float* bg      = (const float*)d_in[9];
    const float* Wout    = (const float*)d_in[10];
    const float* boutp   = (const float*)d_in[11];
    float* out = (float*)d_out;

    char* ws = (char*)d_ws;
    size_t off = 0;
    auto alloc = [&](size_t bytes) {
        size_t p = off;
        off = (off + bytes + 255) & ~(size_t)255;
        return p;
    };
    _Float16* et      = (_Float16*)(ws + alloc((size_t)SEQ * BATCH * EMB * 2));
    _Float16* WpackT  = (_Float16*)(ws + alloc((size_t)NWG * GCOLS * KDIM * 2));
    float*    bpack   = (float*)   (ws + alloc((size_t)NWG * GCOLS * 4));
    _Float16* WoutT   = (_Float16*)(ws + alloc((size_t)VOCAB * HID * 2));
    _Float16* hsbuf   = (_Float16*)(ws + alloc((size_t)SEQ * BATCH * HID * 2));
    _Float16* hzero   = (_Float16*)(ws + alloc((size_t)BATCH * HID * 2));
    unsigned* counter = (unsigned*)(ws + alloc(256));

    k_embed<<<dim3(SEQ * BATCH), dim3(EMB), 0, stream>>>(x, emb, et);
    k_pack<<<dim3(2048), dim3(256), 0, stream>>>(Wf, Wi, Wo, Wg, bf, bi, bo, bg,
                                                 Wout, WpackT, bpack, WoutT,
                                                 hzero, counter);
    k_lstm<<<dim3(NWG), dim3(256), 0, stream>>>(WpackT, bpack, et, hzero,
                                                hsbuf, counter);
    k_proj<<<dim3(SEQ * BATCH / 128, VOCAB / 64), dim3(256), 0, stream>>>(
        hsbuf, WoutT, boutp, out);
}
